// CausalSelfAttention_90761248899419
// MI455X (gfx1250) — matrix operand order repaired and checked by reading, not compiled, not run on hardware
//
#include <hip/hip_runtime.h>

#define Bn 4
#define Sn 2048
#define Dn 1024
#define Hn 16
#define HDn 64

typedef unsigned short u16;
typedef unsigned int   u32;
typedef __attribute__((ext_vector_type(16))) __bf16 v16bf;
typedef __attribute__((ext_vector_type(8)))  float  v8f;
typedef __attribute__((ext_vector_type(4)))  u32    u32x4;
typedef __attribute__((ext_vector_type(4)))  int    i32x4;

#ifndef __has_builtin
#define __has_builtin(x) 0
#endif

#if defined(__AMDGCN__) && __has_builtin(__builtin_amdgcn_global_load_async_to_lds_b128)
#define HAS_ASYNC 1
#else
#define HAS_ASYNC 0
#endif

#define AS_GLOBAL __attribute__((address_space(1)))
#define AS_SHARED __attribute__((address_space(3)))

union Frag { v16bf v; u32x4 q[2]; u16 h[16]; };

__device__ __forceinline__ u16 f2bf(float f) {
  u32 u = __float_as_uint(f);
  u += 0x7FFFu + ((u >> 16) & 1u);           // round-to-nearest-even
  return (u16)(u >> 16);
}

__device__ __forceinline__ v8f wmma_bf16(const Frag& a, const Frag& b, v8f c) {
  return __builtin_amdgcn_wmma_f32_16x16x32_bf16(
      false, a.v, false, b.v, (short)0, c, false, false);
}

// 16-byte global -> LDS stage (async on CDNA5, sync fallback elsewhere)
__device__ __forceinline__ void stage16(const u16* __restrict__ src, u16* dst) {
#if HAS_ASYNC
  __builtin_amdgcn_global_load_async_to_lds_b128(
      (AS_GLOBAL i32x4*)(u16*)src,
      (AS_SHARED i32x4*)dst, 0, 0);
#else
  *(u32x4*)dst = *(const u32x4*)src;
#endif
}

__device__ __forceinline__ void wait_async() {
#if HAS_ASYNC
#if __has_builtin(__builtin_amdgcn_s_wait_asynccnt)
  __builtin_amdgcn_s_wait_asynccnt(0);
#else
  asm volatile("s_wait_asynccnt 0" ::: "memory");
#endif
#endif
}

// ---------------------------------------------------------------- convert
__global__ void cvt_bf16_kernel(const float* __restrict__ in,
                                u16* __restrict__ out, int n) {
  int i = blockIdx.x * blockDim.x + threadIdx.x;
  if (i < n) out[i] = f2bf(in[i]);
}

// ---------------------------------------------------------------- GEMM core
// acc[4] (16x64 per wave) += A[arow,1024] * W[nbase..+64,1024]^T
// B tile staged in double-buffered LDS via async-to-LDS; A double-buffered in regs.
__device__ __forceinline__ void gemm_body(const u16* __restrict__ Ag,
                                          const u16* __restrict__ Wb,
                                          u16* Bl,   // LDS, 2 * 64 * 32 u16
                                          int arow, int nbase, int l16, int hi,
                                          int tid, v8f acc[4]) {
  const int srow   = tid >> 2;          // 0..63  (N row of B tile)
  const int schunk = (tid & 3) * 8;     // 0,8,16,24 (K chunk)
  const u16* wrow = Wb + (size_t)(nbase + srow) * Dn + schunk;

  // prime buffer 0 with K-step 0
  stage16(wrow, Bl + srow * 32 + schunk);
  wait_async();
  __syncthreads();

  Frag a_cur, a_nxt;
  {
    const u16* ap = Ag + (size_t)arow * Dn;
    a_cur.q[0] = *(const u32x4*)(ap + hi * 8);
    a_cur.q[1] = *(const u32x4*)(ap + 16 + hi * 8);
  }

  int buf = 0;
  for (int k0 = 0; k0 < Dn; k0 += 32) {
    const int kn = k0 + 32;
    if (kn < Dn) {
      // async-fill the other LDS buffer + prefetch next A frag into regs
      stage16(wrow + kn, Bl + (buf ^ 1) * (64 * 32) + srow * 32 + schunk);
      const u16* ap = Ag + (size_t)arow * Dn + kn;
      a_nxt.q[0] = *(const u32x4*)(ap + hi * 8);
      a_nxt.q[1] = *(const u32x4*)(ap + 16 + hi * 8);
    }
    const u16* bl = Bl + buf * (64 * 32);
#pragma unroll
    for (int t = 0; t < 4; ++t) {
      Frag b;
      const u16* bp = bl + (t * 16 + l16) * 32;
      b.q[0] = *(const u32x4*)(bp + hi * 8);
      b.q[1] = *(const u32x4*)(bp + 16 + hi * 8);
      acc[t] = wmma_bf16(a_cur, b, acc[t]);
    }
    wait_async();
    __syncthreads();
    a_cur = a_nxt;
    buf ^= 1;
  }
}

// ---------------------------------------------------------------- QKV GEMM
// C[8192,3072] = X * Win^T; scatter Q,K -> [B,H,S,HD]; V -> [B,H,HD,S] (transposed)
__global__ void gemm_qkv_kernel(const u16* __restrict__ xb,
                                const u16* __restrict__ wb,
                                u16* __restrict__ qh,
                                u16* __restrict__ kh,
                                u16* __restrict__ vt) {
  __shared__ __align__(16) u16 Bl[2 * 64 * 32];
  const int wave = threadIdx.x >> 5;
  const int lane = threadIdx.x & 31;
  const int l16  = lane & 15;
  const int hi   = lane >> 4;
  const int mbase = blockIdx.x * 128 + wave * 16;
  const int nbase = blockIdx.y * 64;

  v8f acc[4] = {{}, {}, {}, {}};
  gemm_body(xb, wb, Bl, mbase + l16, nbase, l16, hi, threadIdx.x, acc);

#pragma unroll
  for (int t = 0; t < 4; ++t) {
    int e = nbase + t * 16 + l16;
#pragma unroll
    for (int r = 0; r < 8; ++r) {
      int m  = mbase + r + hi * 8;
      int bb = m >> 11, s = m & (Sn - 1);
      u16 bv = f2bf(acc[t][r]);
      if (e < Dn) {
        int h = e >> 6, hd = e & 63;
        qh[(((bb * Hn + h) * Sn) + s) * HDn + hd] = bv;
      } else if (e < 2 * Dn) {
        int e2 = e - Dn; int h = e2 >> 6, hd = e2 & 63;
        kh[(((bb * Hn + h) * Sn) + s) * HDn + hd] = bv;
      } else {
        int e2 = e - 2 * Dn; int h = e2 >> 6, hd = e2 & 63;
        vt[(((bb * Hn + h) * HDn) + hd) * Sn + s] = bv;
      }
    }
  }
}

// ---------------------------------------------------------------- flash attention
// grid = (S/128, H, B); each wave: 16 query rows, online softmax over 64-key blocks
__global__ void attn_kernel(const u16* __restrict__ qh,
                            const u16* __restrict__ kh,
                            const u16* __restrict__ vt,
                            u16* __restrict__ yb) {
  const int qt = blockIdx.x, h = blockIdx.y, b = blockIdx.z;
  const int wave = threadIdx.x >> 5;
  const int lane = threadIdx.x & 31;
  const int l16  = lane & 15;
  const int hi   = lane >> 4;

  const u16* Q  = qh + (size_t)(b * Hn + h) * Sn * HDn;
  const u16* K  = kh + (size_t)(b * Hn + h) * Sn * HDn;
  const u16* Vt = vt + (size_t)(b * Hn + h) * HDn * Sn;

  const int qbase = qt * 128 + wave * 16;
  const int arow  = qbase + l16;

  __shared__ __align__(16) u16 Pl[8][16][64];

  v8f o[4] = {{}, {}, {}, {}};
  float m_i[8], l_i[8];
#pragma unroll
  for (int r = 0; r < 8; ++r) { m_i[r] = -__builtin_inff(); l_i[r] = 0.f; }

  // preload Q fragments (hd k-steps 0..31, 32..63)
  Frag qa[2];
#pragma unroll
  for (int ks = 0; ks < 2; ++ks) {
    const u16* qp = Q + arow * HDn + ks * 32;
    qa[ks].q[0] = *(const u32x4*)(qp + hi * 8);
    qa[ks].q[1] = *(const u32x4*)(qp + 16 + hi * 8);
  }

  const int nj = qt * 2 + 2;  // key blocks of 64 up to (and incl.) diagonal
  for (int j = 0; j < nj; ++j) {
    const int jbase = j * 64;

    // warm the cache for the V rows this iteration will need after softmax
#pragma unroll
    for (int t = 0; t < 4; ++t)
      __builtin_prefetch(Vt + (t * 16 + l16) * Sn + jbase + hi * 16, 0, 1);

    v8f sacc[4] = {{}, {}, {}, {}};
#pragma unroll
    for (int t = 0; t < 4; ++t) {
#pragma unroll
      for (int ks = 0; ks < 2; ++ks) {
        Frag kb;
        const u16* kp = K + (jbase + t * 16 + l16) * HDn + ks * 32;
        kb.q[0] = *(const u32x4*)(kp + hi * 8);
        kb.q[1] = *(const u32x4*)(kp + 16 + hi * 8);
        sacc[t] = wmma_bf16(qa[ks], kb, sacc[t]);
      }
    }
    // scale, causal mask, row max
    float rmax[8];
#pragma unroll
    for (int r = 0; r < 8; ++r) rmax[r] = -__builtin_inff();
#pragma unroll
    for (int t = 0; t < 4; ++t) {
      int kpcol = jbase + t * 16 + l16;
#pragma unroll
      for (int r = 0; r < 8; ++r) {
        int qq = qbase + r + hi * 8;
        float v = sacc[t][r] * 0.125f;  // 1/sqrt(64)
        if (kpcol > qq) v = -__builtin_inff();
        sacc[t][r] = v;
        rmax[r] = fmaxf(rmax[r], v);
      }
    }
#pragma unroll
    for (int msk = 1; msk < 16; msk <<= 1)
#pragma unroll
      for (int r = 0; r < 8; ++r)
        rmax[r] = fmaxf(rmax[r], __shfl_xor(rmax[r], msk, 32));

    float rsum[8];
#pragma unroll
    for (int r = 0; r < 8; ++r) {
      float mnew  = fmaxf(m_i[r], rmax[r]);
      float alpha = __expf(m_i[r] - mnew);
      m_i[r] = mnew;
      l_i[r] *= alpha;
#pragma unroll
      for (int t = 0; t < 4; ++t) o[t][r] *= alpha;
      float acc = 0.f;
#pragma unroll
      for (int t = 0; t < 4; ++t) {
        float p = __expf(sacc[t][r] - mnew);
        sacc[t][r] = p;
        acc += p;
      }
      rsum[r] = acc;
    }
#pragma unroll
    for (int msk = 1; msk < 16; msk <<= 1)
#pragma unroll
      for (int r = 0; r < 8; ++r) rsum[r] += __shfl_xor(rsum[r], msk, 32);
#pragma unroll
    for (int r = 0; r < 8; ++r) l_i[r] += rsum[r];

    // stage P (C-layout -> LDS row-major) for A-fragment reload
#pragma unroll
    for (int t = 0; t < 4; ++t)
#pragma unroll
      for (int r = 0; r < 8; ++r)
        Pl[wave][r + hi * 8][t * 16 + l16] = f2bf(sacc[t][r]);
    __syncthreads();

    // O += P * V  (V stored transposed -> K-contiguous B fragments)
#pragma unroll
    for (int ks = 0; ks < 2; ++ks) {
      Frag pa;
      const u16* pp = &Pl[wave][l16][ks * 32];
      pa.q[0] = *(const u32x4*)(pp + hi * 8);
      pa.q[1] = *(const u32x4*)(pp + 16 + hi * 8);
#pragma unroll
      for (int t = 0; t < 4; ++t) {
        Frag vb;
        const u16* vp = Vt + (t * 16 + l16) * Sn + jbase + ks * 32;
        vb.q[0] = *(const u32x4*)(vp + hi * 8);
        vb.q[1] = *(const u32x4*)(vp + 16 + hi * 8);
        o[t] = wmma_bf16(pa, vb, o[t]);
      }
    }
    __syncthreads();
  }

  // normalize + store y (recombine heads) as bf16
#pragma unroll
  for (int t = 0; t < 4; ++t)
#pragma unroll
    for (int r = 0; r < 8; ++r) {
      int qq = qbase + r + hi * 8;
      float val = o[t][r] / l_i[r];
      yb[((size_t)(b * Sn + qq)) * Dn + h * HDn + t * 16 + l16] = f2bf(val);
    }
}

// ---------------------------------------------------------------- output GEMM
// out[8192,1024] f32 = Y * Wout^T
__global__ void gemm_out_kernel(const u16* __restrict__ yb,
                                const u16* __restrict__ wb,
                                float* __restrict__ out) {
  __shared__ __align__(16) u16 Bl[2 * 64 * 32];
  const int wave = threadIdx.x >> 5;
  const int lane = threadIdx.x & 31;
  const int l16  = lane & 15;
  const int hi   = lane >> 4;
  const int mbase = blockIdx.x * 128 + wave * 16;
  const int nbase = blockIdx.y * 64;

  v8f acc[4] = {{}, {}, {}, {}};
  gemm_body(yb, wb, Bl, mbase + l16, nbase, l16, hi, threadIdx.x, acc);

#pragma unroll
  for (int t = 0; t < 4; ++t) {
    int e = nbase + t * 16 + l16;
#pragma unroll
    for (int r = 0; r < 8; ++r) {
      int m = mbase + r + hi * 8;
      out[(size_t)m * Dn + e] = acc[t][r];
    }
  }
}

// ---------------------------------------------------------------- launch
extern "C" void kernel_launch(void* const* d_in, const int* in_sizes, int n_in,
                              void* d_out, int out_size, void* d_ws, size_t ws_size,
                              hipStream_t stream) {
  const float* x     = (const float*)d_in[0];
  const float* w_in  = (const float*)d_in[1];
  const float* w_out = (const float*)d_in[2];
  float* out = (float*)d_out;

  char* ws = (char*)d_ws;
  const size_t nX   = (size_t)Bn * Sn * Dn;       // 8388608
  const size_t nWin = (size_t)3 * Dn * Dn;        // 3145728
  const size_t nWo  = (size_t)Dn * Dn;            // 1048576
  const size_t nHd  = (size_t)Bn * Hn * Sn * HDn; // 8388608

  u16* xb   = (u16*)(ws);
  u16* winb = (u16*)(ws + 2 * nX);
  u16* wob  = (u16*)(ws + 2 * (nX + nWin));
  u16* qh   = (u16*)(ws + 2 * (nX + nWin + nWo));
  u16* kh   = (u16*)(ws + 2 * (nX + nWin + nWo + nHd));
  u16* vt   = (u16*)(ws + 2 * (nX + nWin + nWo + 2 * nHd));
  u16* yb   = (u16*)(ws + 2 * (nX + nWin + nWo + 3 * nHd));

  cvt_bf16_kernel<<<(int)((nX   + 255) / 256), 256, 0, stream>>>(x,     xb,   (int)nX);
  cvt_bf16_kernel<<<(int)((nWin + 255) / 256), 256, 0, stream>>>(w_in,  winb, (int)nWin);
  cvt_bf16_kernel<<<(int)((nWo  + 255) / 256), 256, 0, stream>>>(w_out, wob,  (int)nWo);

  gemm_qkv_kernel<<<dim3(64, 48), 256, 0, stream>>>(xb, winb, qh, kh, vt);
  attn_kernel<<<dim3(Sn / 128, Hn, Bn), 256, 0, stream>>>(qh, kh, vt, yb);
  gemm_out_kernel<<<dim3(64, 16), 256, 0, stream>>>(yb, wob, out);
}
